// AttnLayer_80178449482249
// MI455X (gfx1250) — matrix operand order repaired and checked by reading, not compiled, not run on hardware
//
#include <hip/hip_runtime.h>
#include <hip/hip_bf16.h>
#include <math.h>
#include <stdint.h>

// ---------------------------------------------------------------------------
// Problem constants (from the reference): T=8192, XD=4096, RED=8, CS=64
// ---------------------------------------------------------------------------
#define T_TOK 8192
#define XD    4096
#define DK    512      // XD / RED
#define CS    64

typedef __bf16 v16bf __attribute__((ext_vector_type(16)));
typedef float  v8f   __attribute__((ext_vector_type(8)));

union FragBF {
    v16bf v;
    uint4 q[2];
};

__device__ __forceinline__ v8f vzero() {
    v8f z = {0.f, 0.f, 0.f, 0.f, 0.f, 0.f, 0.f, 0.f};
    return z;
}

__device__ __forceinline__ float bf2f(__hip_bfloat16 h) { return __bfloat162float(h); }
__device__ __forceinline__ __hip_bfloat16 f2bf(float f)  { return __float2bfloat16(f); }

// ---------------------------------------------------------------------------
// CDNA5 async copy: GLOBAL_LOAD_ASYNC_TO_LDS_B128 (CDNA5 ISA)
// GV addressing: VDST gives the LDS byte offset, VADDR the 64-bit global address.
// Tracked by ASYNCcnt; loads complete ("done") in order -> s_wait_asynccnt N
// guarantees all but the newest N are in LDS.
// LDS byte offset = low 32 bits of the generic pointer.
// ---------------------------------------------------------------------------
__device__ __forceinline__ void async_copy_b128(const void* lds_dst, const void* gsrc) {
    unsigned           lds = (unsigned)(uintptr_t)lds_dst;
    unsigned long long ga  = (unsigned long long)(uintptr_t)gsrc;
    asm volatile("global_load_async_to_lds_b128 %0, %1, off"
                 :: "v"(lds), "v"(ga)
                 : "memory");
}

template <int N>
__device__ __forceinline__ void s_wait_asynccnt() {
    asm volatile("s_wait_asynccnt %0" :: "n"(N) : "memory");
}

// ---------------------------------------------------------------------------
// WMMA fragment loaders (CDNA5 ISA §7.12.2, 16-bit layouts).
//
// A operand, 16x32 (MxK), source row-major with K contiguous:
//   lane L<16 : row M=L,     VGPR0-3 = K[0..7],  VGPR4-7 = K[16..23]
//   lane L>=16: row M=L-16,  VGPR0-3 = K[8..15], VGPR4-7 = K[24..31]
// -> two contiguous 16B loads per lane.
// ---------------------------------------------------------------------------
__device__ __forceinline__ FragBF load_fragA(const __hip_bfloat16* base, int ld,
                                             int row0, int k0, int lane) {
    FragBF f;
    const int half = lane >> 4;       // 0/1
    const int r    = lane & 15;       // M within tile
    const __hip_bfloat16* p = base + (size_t)(row0 + r) * ld + k0 + (half << 3);
    f.q[0] = *(const uint4*)(p);       // 8 bf16: K k0+8*half .. +7
    f.q[1] = *(const uint4*)(p + 16);  // 8 bf16: K k0+16+8*half .. +7
    return f;
}

// B operand, 32x16 (KxN), source stored (N,K) row-major with K contiguous:
//   lane L<16 : col N=L,     VGPR0-3 = K[0..7],  VGPR4-7 = K[16..23]
//   lane L>=16: col N=L-16,  VGPR0-3 = K[8..15], VGPR4-7 = K[24..31]
// -> two contiguous 16B loads per lane.
__device__ __forceinline__ FragBF load_fragB(const __hip_bfloat16* base, int ld,
                                             int n0, int k0, int lane) {
    FragBF f;
    const int r = lane & 15;                      // N within tile
    const int koff = (lane >> 4) << 3;            // 0 or 8
    const __hip_bfloat16* p = base + (size_t)(n0 + r) * ld + k0 + koff;
    f.q[0] = *(const uint4*)(p);
    f.q[1] = *(const uint4*)(p + 16);
    return f;
}

__device__ __forceinline__ v8f wmma32(FragBF a, FragBF b, v8f c) {
    // D = A(16x32 bf16) * B(32x16 bf16) + C(16x16 f32)
    return __builtin_amdgcn_wmma_f32_16x16x32_bf16(
        /*neg_a=*/false, a.v, /*neg_b=*/false, b.v,
        /*c_mod=*/(short)0, c, /*reuse_a=*/false, /*reuse_b=*/false);
}

// ---------------------------------------------------------------------------
// f32 -> bf16 cast (grid-stride)
// ---------------------------------------------------------------------------
__global__ void cast_f32_bf16(const float* __restrict__ in,
                              __hip_bfloat16* __restrict__ out, long n) {
    long i = (long)blockIdx.x * blockDim.x + threadIdx.x;
    long stride = (long)gridDim.x * blockDim.x;
    for (; i < n; i += stride) out[i] = __float2bfloat16(in[i]);
}

// ---------------------------------------------------------------------------
// GEMM: C(MxN) = A(MxK) * B(NxK)^T, bf16 inputs, f32 WMMA accumulation.
//
// Block = 256 threads = 8 wave32. Block tile 128x128, K-stage 64.
// A/B stages are DMA'd into double-buffered LDS via
// global_load_async_to_lds_b128 (8 x 16B chunks per thread per stage,
// 128B-contiguous per 8 consecutive threads -> fully coalesced), with
// s_wait_asynccnt 8 overlapping stage s+1 DMA with stage s WMMAs.
// LDS tile rows padded to 72 elements (144B) so the 16 rows of a fragment
// hit distinct bank groups on the ds_load_b128s.
// Wave tile 32x64 (2x4 accumulators). Epilogues:
//   0 = store f32, 1 = store bf16, 2 = C = acc * sigmoid(C_prev).
// ---------------------------------------------------------------------------
enum { EPI_F32 = 0, EPI_BF16 = 1, EPI_SIGMUL = 2 };

#define BK   64
#define LDT  72   // padded LDS row length (elements)

__global__ void __launch_bounds__(256) gemm_bf16_nt(
    const __hip_bfloat16* __restrict__ A,
    const __hip_bfloat16* __restrict__ B,
    float* __restrict__ Cf,
    __hip_bfloat16* __restrict__ Cb,
    int M, int N, int K, int epi) {
    __shared__ __align__(16) __hip_bfloat16 sA[2][128][LDT];   // 2 x 18 KB
    __shared__ __align__(16) __hip_bfloat16 sB[2][128][LDT];   // 2 x 18 KB

    const int tid  = threadIdx.x;
    const int lane = tid & 31;
    const int wave = tid >> 5;
    const int rm0 = blockIdx.y * 128;                 // block row base
    const int cn0 = blockIdx.x * 128;                 // block col base
    const int wm = (wave >> 1) * 32;                  // wave row offset in tile
    const int wn = (wave & 1) * 64;                   // wave col offset in tile

    // Issue one K-stage of async tile DMA (A then B; 8 async b128 / thread).
    auto stage_tiles = [&](int k0, int buf) {
#pragma unroll
        for (int i = 0; i < 4; ++i) {
            const int c   = tid + i * 256;            // chunk id 0..1023
            const int row = c >> 3;                   // 8 chunks (128B) per row
            const int cc  = (c & 7) * 8;              // element offset in row
            async_copy_b128(&sA[buf][row][cc],
                            A + (size_t)(rm0 + row) * K + k0 + cc);
        }
#pragma unroll
        for (int i = 0; i < 4; ++i) {
            const int c   = tid + i * 256;
            const int row = c >> 3;
            const int cc  = (c & 7) * 8;
            async_copy_b128(&sB[buf][row][cc],
                            B + (size_t)(cn0 + row) * K + k0 + cc);
        }
    };

    v8f acc[2][4];
#pragma unroll
    for (int i = 0; i < 2; i++)
#pragma unroll
        for (int j = 0; j < 4; j++) acc[i][j] = vzero();

    const int stages = K / BK;
    stage_tiles(0, 0);

    for (int s = 0; s < stages; ++s) {
        const int cur = s & 1;
        if (s + 1 < stages) {
            stage_tiles((s + 1) * BK, cur ^ 1);   // prefetch next stage
            s_wait_asynccnt<8>();                 // stage s (all but newest 8) landed
        } else {
            s_wait_asynccnt<0>();
        }
        __syncthreads();                          // stage s visible to all waves

#pragma unroll
        for (int kk = 0; kk < BK; kk += 32) {
            FragBF aF[2], bF[4];
#pragma unroll
            for (int i = 0; i < 2; i++)
                aF[i] = load_fragA(&sA[cur][0][0], LDT, wm + i * 16, kk, lane);
#pragma unroll
            for (int j = 0; j < 4; j++)
                bF[j] = load_fragB(&sB[cur][0][0], LDT, wn + j * 16, kk, lane);
#pragma unroll
            for (int i = 0; i < 2; i++)
#pragma unroll
                for (int j = 0; j < 4; j++)
                    acc[i][j] = wmma32(aF[i], bF[j], acc[i][j]);
        }
        __syncthreads();                          // done reading buf before overwrite
    }

    // C/D layout: VGPR r -> M = r (+8 for lanes 16..31), N = lane&15
    const int half = lane >> 4;
    const int c0   = lane & 15;
#pragma unroll
    for (int i = 0; i < 2; i++) {
#pragma unroll
        for (int j = 0; j < 4; j++) {
#pragma unroll
            for (int r = 0; r < 8; r++) {
                const int row = rm0 + wm + i * 16 + half * 8 + r;
                const int col = cn0 + wn + j * 16 + c0;
                const size_t idx = (size_t)row * N + col;
                const float v = acc[i][j][r];
                if (epi == EPI_BF16) {
                    Cb[idx] = f2bf(v);
                } else if (epi == EPI_F32) {
                    Cf[idx] = v;
                } else {
                    const float sg = 1.f / (1.f + __expf(-Cf[idx]));
                    Cf[idx] = v * sg;
                }
            }
        }
    }
}

// ---------------------------------------------------------------------------
// Attention over one chunk: 64 queries x 128 keys (prev chunk + current,
// zero-padded window for chunk 0), RoPE, masked softmax, A@V.
// One block (8 wave32) per chunk; q'/k' staged in LDS after RoPE; qk and A@V
// done with V_WMMA_F32_16X16X32_BF16; V tiles transposed through LDS.
// ---------------------------------------------------------------------------
__global__ void __launch_bounds__(256) attn_chunk(
    const __hip_bfloat16* __restrict__ Q,   // (8192, 512)  qs
    const __hip_bfloat16* __restrict__ Kx,  // (8192, 512)  ksx
    const __hip_bfloat16* __restrict__ V,   // (8192, 4096) vsx
    __hip_bfloat16* __restrict__ Y) {       // (8192, 4096) ys
    __shared__ __align__(16) __hip_bfloat16 sQ[64 * 512];    //  64 KB  RoPE'd q
    __shared__ __align__(16) __hip_bfloat16 sK[128 * 512];   // 128 KB  RoPE'd k window
    __shared__ __align__(16) float          sS[64 * 128];    //  32 KB  raw scores
    __shared__ __align__(16) __hip_bfloat16 sA[64 * 128];    //  16 KB  softmax probs
    __shared__ __align__(16) __hip_bfloat16 sV[64 * 136];    //  17 KB  v^T tile (n,k), padded

    const int ci   = blockIdx.x;
    const int tid  = threadIdx.x;
    const int lane = tid & 31;
    const int wave = tid >> 5;
    const float LOG1E4 = 9.210340371976184f;   // ln(10000)

    // ---- Phase 1: RoPE q (pos 64..127) and k window (pos 0..127) into LDS
    for (int it = tid; it < (64 + 128) * 256; it += 256) {
        const int  isK = it >= 64 * 256;
        const int  idx = isK ? it - 64 * 256 : it;
        const int  row = idx >> 8;       // q: 0..63, k: 0..127
        const int  p   = idx & 255;      // rotary pair index (d/2 = 256)
        const float freq = __expf((float)p * (-2.0f / 512.0f) * LOG1E4);
        float x0 = 0.f, x1 = 0.f, pos;
        if (!isK) {
            const long g = (long)ci * 64 + row;
            x0 = bf2f(Q[g * 512 + 2 * p]);
            x1 = bf2f(Q[g * 512 + 2 * p + 1]);
            pos = (float)(64 + row);
        } else {
            const long g = (long)(ci - 1) * 64 + row;   // window row r -> global row
            if (g >= 0) {
                x0 = bf2f(Kx[g * 512 + 2 * p]);
                x1 = bf2f(Kx[g * 512 + 2 * p + 1]);
            }
            pos = (float)row;
        }
        float s, c;
        __sincosf(pos * freq, &s, &c);
        const float y0 = x0 * c - x1 * s;   // even lane of pair
        const float y1 = x1 * c + x0 * s;   // odd lane of pair
        if (!isK) { sQ[row * 512 + 2 * p] = f2bf(y0); sQ[row * 512 + 2 * p + 1] = f2bf(y1); }
        else      { sK[row * 512 + 2 * p] = f2bf(y0); sK[row * 512 + 2 * p + 1] = f2bf(y1); }
    }
    __syncthreads();

    // ---- Phase 2: qk (64x128, K=512) via WMMA. 32 tiles; 4 per wave.
    {
        const int mt  = wave >> 1;          // 0..3
        const int ntb = (wave & 1) * 4;     // 0 or 4
        v8f acc[4];
#pragma unroll
        for (int t = 0; t < 4; t++) acc[t] = vzero();
        for (int kk = 0; kk < 512; kk += 32) {
            const FragBF aF = load_fragA(sQ, 512, mt * 16, kk, lane);
#pragma unroll
            for (int t = 0; t < 4; t++) {
                const FragBF bF = load_fragB(sK, 512, (ntb + t) * 16, kk, lane);
                acc[t] = wmma32(aF, bF, acc[t]);
            }
        }
        const int half = lane >> 4, c0 = lane & 15;
#pragma unroll
        for (int t = 0; t < 4; t++)
#pragma unroll
            for (int r = 0; r < 8; r++)
                sS[(mt * 16 + half * 8 + r) * 128 + (ntb + t) * 16 + c0] = acc[t][r];
    }
    __syncthreads();

    // ---- Phase 3: masked softmax per row (mask: m <= j+64; zero-pad keys
    //      legitimately participate with score 0, matching the reference).
    if (tid < 64) {
        const float scale = 0.044194173824159216f;   // 1/sqrt(512)
        const int j = tid, lim = j + 64;
        float mx = -1e30f;
        for (int m = 0; m <= lim; m++) mx = fmaxf(mx, sS[j * 128 + m] * scale);
        float sum = 0.f;
        for (int m = 0; m <= lim; m++) sum += __expf(sS[j * 128 + m] * scale - mx);
        const float inv = 1.f / sum;
        for (int m = 0; m < 128; m++) {
            const float a = (m <= lim) ? __expf(sS[j * 128 + m] * scale - mx) * inv : 0.f;
            sA[j * 128 + m] = f2bf(a);
        }
    }
    __syncthreads();

    // ---- Phase 4: ys(64 x 4096) = A(64x128) @ Vwin(128x4096); V tiles of 64
    //      columns transposed into LDS so the B operand is K-contiguous.
    for (int nb = 0; nb < 64; nb++) {
        const int n0 = nb * 64;
        for (int it = tid; it < 64 * 128; it += 256) {
            const int k = it >> 6, n = it & 63;   // consecutive tid -> consecutive n (coalesced)
            const long g = (long)(ci - 1) * 64 + k;
            sV[n * 136 + k] = (g >= 0) ? V[g * 4096 + n0 + n] : f2bf(0.f);
        }
        __syncthreads();
#pragma unroll
        for (int t = 0; t < 2; t++) {
            const int tile = wave * 2 + t;          // 16 tiles: 4 (M) x 4 (N)
            const int mt = tile >> 2, nt = tile & 3;
            v8f acc = vzero();
#pragma unroll
            for (int kk = 0; kk < 128; kk += 32) {
                const FragBF aF = load_fragA(sA, 128, mt * 16, kk, lane);
                const FragBF bF = load_fragB(sV, 136, nt * 16, kk, lane);
                acc = wmma32(aF, bF, acc);
            }
            const int half = lane >> 4, c0 = lane & 15;
#pragma unroll
            for (int r = 0; r < 8; r++) {
                const long row = (long)ci * 64 + mt * 16 + half * 8 + r;
                const int  col = n0 + nt * 16 + c0;
                Y[row * 4096 + col] = f2bf(acc[r]);
            }
        }
        __syncthreads();
    }
}

// ---------------------------------------------------------------------------
// Host-side orchestration (all on `stream`, graph-capture safe).
// ---------------------------------------------------------------------------
extern "C" void kernel_launch(void* const* d_in, const int* in_sizes, int n_in,
                              void* d_out, int out_size, void* d_ws, size_t ws_size,
                              hipStream_t stream) {
    const float* xs = (const float*)d_in[0];
    const float* Wq = (const float*)d_in[1];
    const float* Wk = (const float*)d_in[2];
    const float* Wv = (const float*)d_in[3];
    const float* Wo = (const float*)d_in[4];
    const float* Wr = (const float*)d_in[5];
    float* out = (float*)d_out;
    (void)in_sizes; (void)n_in; (void)out_size; (void)ws_size;

    // Workspace carve-up (bf16 buffers, ~309 MiB total)
    char* ws = (char*)d_ws;
    size_t off = 0;
    auto carve = [&](size_t elems) -> __hip_bfloat16* {
        __hip_bfloat16* p = (__hip_bfloat16*)(ws + off);
        off += ((elems * sizeof(__hip_bfloat16)) + 255) & ~(size_t)255;
        return p;
    };
    __hip_bfloat16* xb  = carve((size_t)T_TOK * XD);   // xs        bf16
    __hip_bfloat16* wqb = carve((size_t)DK * XD);      // Wq        bf16
    __hip_bfloat16* wkb = carve((size_t)DK * DK);      // Wk        bf16
    __hip_bfloat16* wvb = carve((size_t)XD * XD);      // Wv        bf16
    __hip_bfloat16* wob = carve((size_t)XD * XD);      // Wo        bf16
    __hip_bfloat16* wrb = carve((size_t)XD * XD);      // Wr        bf16
    __hip_bfloat16* qb  = carve((size_t)T_TOK * DK);   // qs        bf16
    __hip_bfloat16* kb  = carve((size_t)T_TOK * DK);   // ksx       bf16
    __hip_bfloat16* vb  = carve((size_t)T_TOK * XD);   // vsx       bf16
    __hip_bfloat16* ysb = carve((size_t)T_TOK * XD);   // ys        bf16

    const dim3 blk(256);

    // 0) f32 -> bf16 casts
    cast_f32_bf16<<<2048, blk, 0, stream>>>(xs, xb, (long)T_TOK * XD);
    cast_f32_bf16<<<256,  blk, 0, stream>>>(Wq, wqb, (long)DK * XD);
    cast_f32_bf16<<<64,   blk, 0, stream>>>(Wk, wkb, (long)DK * DK);
    cast_f32_bf16<<<1024, blk, 0, stream>>>(Wv, wvb, (long)XD * XD);
    cast_f32_bf16<<<1024, blk, 0, stream>>>(Wo, wob, (long)XD * XD);
    cast_f32_bf16<<<1024, blk, 0, stream>>>(Wr, wrb, (long)XD * XD);

    // 1) qs = xs @ Wq^T            (8192 x 512 x 4096)  -> bf16
    gemm_bf16_nt<<<dim3(DK / 128, T_TOK / 128), blk, 0, stream>>>(
        xb, wqb, nullptr, qb, T_TOK, DK, XD, EPI_BF16);
    // 2) ksx = qs @ Wk^T           (8192 x 512 x 512)   -> bf16
    gemm_bf16_nt<<<dim3(DK / 128, T_TOK / 128), blk, 0, stream>>>(
        qb, wkb, nullptr, kb, T_TOK, DK, DK, EPI_BF16);
    // 3) vsx = xs @ Wv^T           (8192 x 4096 x 4096) -> bf16
    gemm_bf16_nt<<<dim3(XD / 128, T_TOK / 128), blk, 0, stream>>>(
        xb, wvb, nullptr, vb, T_TOK, XD, XD, EPI_BF16);

    // 4) chunked windowed attention (RoPE + softmax + A@V), one block/chunk
    attn_chunk<<<T_TOK / CS, blk, 0, stream>>>(qb, kb, vb, ysb);

    // 5) r = xs @ Wr^T -> f32 parked in d_out
    gemm_bf16_nt<<<dim3(XD / 128, T_TOK / 128), blk, 0, stream>>>(
        xb, wrb, out, nullptr, T_TOK, XD, XD, EPI_F32);
    // 6) out = (ys @ Wo^T) * sigmoid(r)   (fused epilogue reads/overwrites d_out)
    gemm_bf16_nt<<<dim3(XD / 128, T_TOK / 128), blk, 0, stream>>>(
        ysb, wob, out, nullptr, T_TOK, XD, XD, EPI_SIGMUL);
}
